// DynamicGatedAttention_65936337928202
// MI455X (gfx1250) — matrix operand order repaired and checked by reading, not compiled, not run on hardware
//
#include <hip/hip_runtime.h>

// ---------------- problem constants (match reference) ----------------
#define NC 128   // channels C
#define NG 8     // groups G
#define NK 16    // neighbors K
#define SCALE_F 2.0f
#define BASE_F 10000.0f

// ---------------- misc (stats) buffer layout, float indices ----------
#define QK_SUM 0      // 256: sum for q(0..127) and k(128..255) pre-BN
#define QK_SQ  256    // 256
#define H_SUM  512    // 128: gate hidden
#define H_SQ   640    // 128
#define G_SUM  768    // 8  : geom hidden
#define G_SQ   776    // 8
#define GMAX   784    // 512: per-segment channel max of q_raw (B<=4)
#define ZBIAS  1296   // 512: zero bias for rope GEMMs
#define QKVB   1808   // 384: concat(bq, bk, bv)
#define MISC_FLOATS 4096

typedef __attribute__((ext_vector_type(16))) __bf16 v16bf;
typedef __attribute__((ext_vector_type(8)))  float  v8f;

__device__ __forceinline__ unsigned short f2bf(float x) {
  unsigned int u = __float_as_uint(x);
  if ((u & 0x7f800000u) != 0x7f800000u)        // finite: round-to-nearest-even
    u += 0x7fffu + ((u >> 16) & 1u);
  return (unsigned short)(u >> 16);
}

__device__ __forceinline__ int seg_of(int n, const int* __restrict__ off, int B) {
  int s = 0;
  for (int b = 0; b < B; ++b) s += (off[b] <= n) ? 1 : 0;   // searchsorted right
  return s;
}

// ---------------- init: zero stats/gmax/zbias, build qkv bias --------
__global__ void init_misc(float* misc, const float* bq, const float* bk, const float* bv) {
  int t = blockIdx.x * blockDim.x + threadIdx.x;
  if (t < QKVB) misc[t] = 0.0f;
  if (t < 3 * NC)
    misc[QKVB + t] = (t < NC) ? bq[t] : (t < 2 * NC ? bk[t - NC] : bv[t - 2 * NC]);
}

// ---------------- weight transpose + bf16 convert --------------------
// WqkvT[n][k] = W*(k, n) for n in [0,384); Wg1T[n][k] = Wg1[k][n], k in [0,256)
__global__ void convert_weights(const float* __restrict__ Wq, const float* __restrict__ Wk,
                                const float* __restrict__ Wv, const float* __restrict__ Wg1,
                                unsigned short* __restrict__ WqkvT,
                                unsigned short* __restrict__ Wg1T) {
  int t = blockIdx.x * blockDim.x + threadIdx.x;
  const int t1 = 3 * NC * NC;
  const int t2 = NC * 2 * NC;
  if (t < t1) {
    int n = t / NC, k = t % NC;
    float w = (n < NC) ? Wq[k * NC + n]
            : (n < 2 * NC) ? Wk[k * NC + (n - NC)] : Wv[k * NC + (n - 2 * NC)];
    WqkvT[t] = f2bf(w);
  } else if (t < t1 + t2) {
    int i = t - t1;
    int n = i / (2 * NC), k = i % (2 * NC);
    Wg1T[i] = f2bf(Wg1[k * NC + n]);
  }
}

// ---------------- generic f32 -> bf16 --------------------------------
__global__ void to_bf16(const float* __restrict__ x, unsigned short* __restrict__ y, int count) {
  int t = blockIdx.x * blockDim.x + threadIdx.x;
  if (t < count) y[t] = f2bf(x[t]);
}

// ---------------- Cayley P = (I+S)^-1 (I-S), Gauss-Jordan ------------
// Single block, 256 threads; augmented [I+Se | I-Se] lives in ws (G).
__global__ void cayley_build_p(const float* __restrict__ S, float* __restrict__ G,
                               float* __restrict__ P, unsigned short* __restrict__ Pbf,
                               unsigned short* __restrict__ PTbf) {
  const int tid = threadIdx.x;
  for (int i = tid; i < NC * 2 * NC; i += 256) {
    int r = i / (2 * NC), c = i % (2 * NC);
    int cc = (c < NC) ? c : c - NC;
    float se = (cc > r ? S[r * NC + cc] : 0.f) - (r > cc ? S[cc * NC + r] : 0.f);
    G[i] = (r == cc ? 1.f : 0.f) + (c < NC ? se : -se);
  }
  __shared__ float spiv[2 * NC];
  for (int i = 0; i < NC; ++i) {
    __syncthreads();
    spiv[tid] = G[i * 2 * NC + tid];
    __syncthreads();
    float inv = 1.0f / spiv[i];
    float pr = spiv[tid] * inv;
    __syncthreads();
    spiv[tid] = pr;
    G[i * 2 * NC + tid] = pr;
    __syncthreads();
    int r = tid >> 1;
    int c0 = (tid & 1) << 7;            // 0 or 128
    float f = G[r * 2 * NC + i];
    __syncthreads();
    if (r != i)
      for (int c = c0; c < c0 + NC; ++c)
        G[r * 2 * NC + c] -= f * spiv[c];
  }
  __syncthreads();
  for (int i = tid; i < NC * NC; i += 256) {
    int r = i / NC, c = i % NC;
    float p = G[r * 2 * NC + NC + c];
    P[i] = p;
    Pbf[i] = f2bf(p);            // WT for  x @ P^T   (WT[n][k] = P[n][k])
    PTbf[c * NC + r] = f2bf(p);  // WT for  y @ P     (WT[n][k] = P[k][n])
  }
}

// ---------------- bf16 WMMA GEMM: out[M][128*TILES] = A[M][KD] @ W + bias
// WT pre-transposed [Nout][KD] bf16. 8 waves/block, 16 rows/block.
// K-step outer loop: A fragment (2x ds_load_b128) reused across TILES n-tiles.
template <int KD, int TILES>
__global__ __launch_bounds__(256)
void wmma_gemm_bf16(const unsigned short* __restrict__ A, const unsigned short* __restrict__ WT,
                    const float* __restrict__ bias, float* __restrict__ out) {
  constexpr int NOUT = 128 * TILES;
  __shared__ unsigned short sA[16 * KD];
  const int tid = threadIdx.x;
  const int row0 = blockIdx.x * 16;
  // stage 16 contiguous rows of A into LDS as uint4s
  const uint4* A128 = (const uint4*)(A + (size_t)row0 * KD);
  uint4* sAv = (uint4*)sA;
  for (int i = tid; i < (16 * KD) / 8; i += 256) sAv[i] = A128[i];
  __syncthreads();

  const int lane = tid & 31, wave = tid >> 5;
  const int half = lane >> 4, lrow = lane & 15;

  int ncol[TILES];
  v8f acc[TILES];
#pragma unroll
  for (int t = 0; t < TILES; ++t) {
    ncol[t] = ((wave + 8 * t) << 4) + lrow;          // column (ISA: N = lane%16)
    float bz = bias[ncol[t]];
#pragma unroll
    for (int r = 0; r < 8; ++r) acc[t][r] = bz;
  }

#pragma unroll
  for (int kk = 0; kk < KD; kk += 32) {
    // A frag: 16 contiguous k values per lane (k = 16*half + s), matching B
    union { v16bf v; uint4 q[2]; } a;
    const int sbase = lrow * KD + kk + 16 * half;    // ushort index, 8-aligned
    a.q[0] = sAv[sbase >> 3];
    a.q[1] = sAv[(sbase + 8) >> 3];
#pragma unroll
    for (int t = 0; t < TILES; ++t) {
      // B frag: 16 contiguous k values of column ncol[t]
      union { v16bf v; uint4 q[2]; } b;
      const uint4* W128 = (const uint4*)(WT + (size_t)ncol[t] * KD + kk + (half << 4));
      b.q[0] = W128[0];
      b.q[1] = W128[1];
      acc[t] = __builtin_amdgcn_wmma_f32_16x16x32_bf16(false, a.v, false, b.v,
                                                       (short)0, acc[t], false, false);
    }
  }

#pragma unroll
  for (int t = 0; t < TILES; ++t) {
    float* orow = out + (size_t)(row0 + (half << 3)) * NOUT + ncol[t];  // M = r + 8*half
#pragma unroll
    for (int r = 0; r < 8; ++r) orow[(size_t)r * NOUT] = acc[t][r];
  }
}

// ---------------- BN stats: per-column sum / sumsq -------------------
__global__ void bn_stats(const float* __restrict__ X, int rows, int ld, int cols,
                         float* __restrict__ sum, float* __restrict__ sq) {
  int tid = threadIdx.x;
  int col = tid % cols;
  int rpb = blockDim.x / cols;
  int r = blockIdx.x * rpb + tid / cols;
  int stride = gridDim.x * rpb;
  float s = 0.f, s2 = 0.f;
  for (; r < rows; r += stride) {
    float v = X[(size_t)r * ld + col];
    s += v; s2 += v * v;
  }
  atomicAdd(&sum[col], s);
  atomicAdd(&sq[col], s2);
}

// ---------------- BN+ReLU for q/k, bf16 out, gmax atomics ------------
__global__ void bn_apply_qk(const float* __restrict__ qkv, float* __restrict__ misc,
                            const float* gq, const float* bnq,
                            const float* gk, const float* bnk,
                            unsigned short* __restrict__ qraw, unsigned short* __restrict__ kraw,
                            const int* __restrict__ off, int B, int N) {
  int t = blockIdx.x * blockDim.x + threadIdx.x;
  if (t >= N * 2 * NC) return;
  int n = t >> 8, c = t & 255;
  float invN = 1.0f / (float)N;
  float mean = misc[QK_SUM + c] * invN;
  float var  = misc[QK_SQ + c] * invN - mean * mean;
  float x = qkv[(size_t)n * (3 * NC) + c];
  float gm = (c < NC) ? gq[c] : gk[c - NC];
  float bt = (c < NC) ? bnq[c] : bnk[c - NC];
  float y = fmaxf((x - mean) * rsqrtf(var + 1e-5f) * gm + bt, 0.0f);
  if (c < NC) {
    qraw[(size_t)n * NC + c] = f2bf(y);
    int s = seg_of(n, off, B);
    atomicMax((int*)&misc[GMAX + s * NC + c], __float_as_int(y));  // y >= 0
  } else {
    kraw[(size_t)n * NC + (c - NC)] = f2bf(y);
  }
}

// ---------------- RoPE rotation (dpa=42, 3 axes, dims 126/127 pass) --
__global__ void rope_rotate(const float* __restrict__ xr, const float* __restrict__ coord,
                            unsigned short* __restrict__ rot, int N) {
  int t = blockIdx.x * blockDim.x + threadIdx.x;
  if (t >= N * 64) return;
  int n = t >> 6, j = t & 63;
  const float* x = xr + (size_t)n * NC;
  unsigned short* o = rot + (size_t)n * NC;
  if (j == 63) { o[126] = f2bf(x[126]); o[127] = f2bf(x[127]); return; }
  int a = j / 21, jj = j % 21;
  int base = a * 42;
  float invf = __powf(BASE_F, -(float)(2 * jj) / 42.0f);
  float ang = coord[n * 3 + a] * SCALE_F * invf;
  float s, co; __sincosf(ang, &s, &co);
  float fr = x[base + jj], fi = x[base + 21 + jj];
  o[base + jj]      = f2bf(fr * co - fi * s);
  o[base + 21 + jj] = f2bf(fr * s + fi * co);
}

// ---------------- gate_in = concat(q_raw, gmax[seg]) in bf16 ---------
__global__ void gate_in_build(const unsigned short* __restrict__ qraw,
                              const float* __restrict__ misc,
                              const int* __restrict__ off, int B,
                              unsigned short* __restrict__ gin, int N) {
  int t = blockIdx.x * blockDim.x + threadIdx.x;
  if (t >= N * 2 * NC) return;
  int n = t >> 8, c = t & 255;
  if (c < NC) gin[t] = qraw[(size_t)n * NC + c];
  else {
    int s = seg_of(n, off, B);
    gin[t] = f2bf(misc[GMAX + s * NC + (c - NC)]);
  }
}

// ---------------- gate = sigmoid(relu(BN(h)) @ Wg2 + bg2) ------------
__global__ void gate_final(const float* __restrict__ h, const float* __restrict__ misc,
                           const float* gg, const float* bng,
                           const float* __restrict__ Wg2, const float* bg2,
                           float* __restrict__ gate, int N) {
  __shared__ float sc[NC], sh[NC];
  int tid = threadIdx.x;
  if (tid < NC) {
    float invN = 1.0f / (float)N;
    float mean = misc[H_SUM + tid] * invN;
    float var  = misc[H_SQ + tid] * invN - mean * mean;
    float s = gg[tid] * rsqrtf(var + 1e-5f);
    sc[tid] = s;
    sh[tid] = bng[tid] - mean * s;
  }
  __syncthreads();
  int t = blockIdx.x * blockDim.x + tid;
  int n = t >> 3, g = t & 7;
  if (n >= N) return;
  const float* hr = h + (size_t)n * NC;
  float acc = bg2[g];
  for (int c = 0; c < NC; ++c) {
    float y = fmaxf(hr[c] * sc[c] + sh[c], 0.0f);
    acc += y * Wg2[c * NG + g];
  }
  gate[(size_t)n * NG + g] = 1.0f / (1.0f + expf(-acc));
}

// ---------------- geom pass1: h2 = (k[ref]-q) @ Ww1 + bw1, + stats ---
__global__ void geom_pass1(const float* __restrict__ qrope, const float* __restrict__ krope,
                           const int* __restrict__ ref, const float* __restrict__ Ww1,
                           const float* __restrict__ bw1, float* __restrict__ h2,
                           float* __restrict__ misc, int N) {
  __shared__ float bs[NG], b2[NG];
  int tid = threadIdx.x;
  if (tid < NG) { bs[tid] = 0.f; b2[tid] = 0.f; }
  __syncthreads();
  int t = blockIdx.x * blockDim.x + tid;
  int n = t / (NK * NG);
  int k = (t / NG) % NK;
  int g = t % NG;
  float acc = 0.f;
  if (n < N) {
    int r = ref[n * NK + k];
    const float* kr = krope + (size_t)r * NC;
    const float* qr = qrope + (size_t)n * NC;
    acc = bw1[g];
    for (int c = 0; c < NC; ++c) acc += (kr[c] - qr[c]) * Ww1[c * NG + g];
    h2[t] = acc;
  }
  atomicAdd(&bs[g], acc);
  atomicAdd(&b2[g], acc * acc);
  __syncthreads();
  if (tid < NG) {
    atomicAdd(&misc[G_SUM + tid], bs[tid]);
    atomicAdd(&misc[G_SQ + tid], b2[tid]);
  }
}

// ---------------- fused tail: BN+relu, @Ww2, gate, softmax, einsum ---
__global__ void geom_final(const float* __restrict__ h2, const float* __restrict__ misc,
                           const float* gw, const float* bnw,
                           const float* __restrict__ Ww2, const float* bw2,
                           const float* __restrict__ gate, const int* __restrict__ ref,
                           const float* __restrict__ qkv, float* __restrict__ out, int N) {
  __shared__ float s_hb[NK][NG], s_rf[NK][NG], s_at[NK][NG];
  __shared__ float s_w2[NG * NG], s_gt[NG];
  __shared__ int s_ri[NK];
  int tid = threadIdx.x;           // 128 threads
  int n = blockIdx.x;
  if (tid < NG * NG) s_w2[tid] = Ww2[tid];
  if (tid < NG) s_gt[tid] = gate[(size_t)n * NG + tid];
  if (tid < NK) {
    int r = ref[n * NK + tid];
    s_ri[tid] = r;
    __builtin_prefetch(&qkv[(size_t)r * (3 * NC) + 2 * NC], 0, 3);  // value row
  }
  int k = tid >> 3, g = tid & 7;
  float cnt = (float)N * (float)NK;
  float mean = misc[G_SUM + g] / cnt;
  float var  = misc[G_SQ + g] / cnt - mean * mean;
  float x = h2[(size_t)n * (NK * NG) + tid];
  float hb = fmaxf((x - mean) * rsqrtf(var + 1e-5f) * gw[g] + bnw[g], 0.0f);
  s_hb[k][g] = hb;
  __syncthreads();
  float geo = bw2[g];
#pragma unroll
  for (int g2 = 0; g2 < NG; ++g2) geo += s_hb[k][g2] * s_w2[g2 * NG + g];
  s_rf[k][g] = geo * (1.0f + s_gt[g]);
  __syncthreads();
  if (tid < NG) {
    float m = -1e30f;
    for (int kk = 0; kk < NK; ++kk) m = fmaxf(m, s_rf[kk][tid]);
    float den = 0.f;
    for (int kk = 0; kk < NK; ++kk) { float e = expf(s_rf[kk][tid] - m); s_at[kk][tid] = e; den += e; }
    float inv = 1.0f / den;
    for (int kk = 0; kk < NK; ++kk) {
      int rp1 = s_ri[kk] + 1;
      float msk = (rp1 > 0) ? 1.f : ((rp1 < 0) ? -1.f : 0.f);
      s_at[kk][tid] *= inv * msk;
    }
  }
  __syncthreads();
  int c = tid;                     // output channel
  int gc = c >> 4;                 // vg4 reshape: g = c / (C/G)
  float acc = 0.f;
  for (int kk = 0; kk < NK; ++kk) {
    int r = s_ri[kk];
    acc += qkv[(size_t)r * (3 * NC) + 2 * NC + c] * s_at[kk][gc];
  }
  out[(size_t)n * NC + c] = acc;
}

// ====================================================================
extern "C" void kernel_launch(void* const* d_in, const int* in_sizes, int n_in,
                              void* d_out, int out_size, void* d_ws, size_t ws_size,
                              hipStream_t stream) {
  const float* feat  = (const float*)d_in[0];
  const float* coord = (const float*)d_in[1];
  const int* refidx  = (const int*)d_in[2];
  const int* offset  = (const int*)d_in[3];
  const float* Wq  = (const float*)d_in[4];
  const float* bq  = (const float*)d_in[5];
  const float* gq  = (const float*)d_in[6];
  const float* bnq = (const float*)d_in[7];
  const float* Wk  = (const float*)d_in[8];
  const float* bk  = (const float*)d_in[9];
  const float* gk  = (const float*)d_in[10];
  const float* bnk = (const float*)d_in[11];
  const float* Wv  = (const float*)d_in[12];
  const float* bv  = (const float*)d_in[13];
  const float* S   = (const float*)d_in[14];
  const float* Ww1 = (const float*)d_in[15];
  const float* bw1 = (const float*)d_in[16];
  const float* gw  = (const float*)d_in[17];
  const float* bnw = (const float*)d_in[18];
  const float* Ww2 = (const float*)d_in[19];
  const float* bw2 = (const float*)d_in[20];
  const float* Wg1 = (const float*)d_in[21];
  const float* bg1 = (const float*)d_in[22];
  const float* gg  = (const float*)d_in[23];
  const float* bng = (const float*)d_in[24];
  const float* Wg2 = (const float*)d_in[25];
  const float* bg2 = (const float*)d_in[26];

  const int N = in_sizes[0] / NC;
  const int B = in_sizes[3];
  (void)n_in; (void)out_size; (void)ws_size;

  // ---- workspace layout (256B aligned slices) ----
  char* ws = (char*)d_ws;
  size_t o = 0;
  auto take = [&](size_t bytes) {
    char* p = ws + o;
    o = (o + bytes + 255) & ~(size_t)255;
    return p;
  };
  float*          misc  = (float*)take(MISC_FLOATS * 4);
  float*          gaussG= (float*)take((size_t)NC * 2 * NC * 4);
  unsigned short* WqkvT = (unsigned short*)take((size_t)3 * NC * NC * 2);
  float*          Pf    = (float*)take((size_t)NC * NC * 4);
  unsigned short* Pbf   = (unsigned short*)take((size_t)NC * NC * 2);
  unsigned short* PTbf  = (unsigned short*)take((size_t)NC * NC * 2);
  unsigned short* Wg1T  = (unsigned short*)take((size_t)NC * 2 * NC * 2);
  unsigned short* featb = (unsigned short*)take((size_t)N * NC * 2);
  float*          qkv   = (float*)take((size_t)N * 3 * NC * 4);
  unsigned short* qraw  = (unsigned short*)take((size_t)N * NC * 2);
  unsigned short* kraw  = (unsigned short*)take((size_t)N * NC * 2);
  float*          xr    = (float*)take((size_t)N * NC * 4);
  unsigned short* rotb  = (unsigned short*)take((size_t)N * NC * 2);
  float*          qrope = (float*)take((size_t)N * NC * 4);
  float*          krope = (float*)take((size_t)N * NC * 4);
  unsigned short* ginb  = (unsigned short*)take((size_t)N * 2 * NC * 2);
  float*          hbuf  = (float*)take((size_t)N * NC * 4);   // gate hidden, then h2
  float*          gatev = (float*)take((size_t)N * NG * 4);

  dim3 blk(256);
  init_misc<<<9, blk, 0, stream>>>(misc, bq, bk, bv);
  convert_weights<<<(3 * NC * NC + 2 * NC * NC + 255) / 256, blk, 0, stream>>>(
      Wq, Wk, Wv, Wg1, WqkvT, Wg1T);
  cayley_build_p<<<1, blk, 0, stream>>>(S, gaussG, Pf, Pbf, PTbf);
  to_bf16<<<(N * NC + 255) / 256, blk, 0, stream>>>(feat, featb, N * NC);

  // qkv = feat @ [Wq|Wk|Wv] + [bq|bk|bv]   (KD=128, 3 tiles/wave -> Nout=384)
  wmma_gemm_bf16<128, 3><<<N / 16, blk, 0, stream>>>(featb, WqkvT, misc + QKVB, qkv);
  bn_stats<<<256, blk, 0, stream>>>(qkv, N, 3 * NC, 2 * NC, misc + QK_SUM, misc + QK_SQ);
  bn_apply_qk<<<(N * 2 * NC) / 256, blk, 0, stream>>>(qkv, misc, gq, bnq, gk, bnk,
                                                      qraw, kraw, offset, B, N);
  // q = rot(q_raw @ P^T) @ P
  wmma_gemm_bf16<128, 1><<<N / 16, blk, 0, stream>>>(qraw, Pbf, misc + ZBIAS, xr);
  rope_rotate<<<(N * 64) / 256, blk, 0, stream>>>(xr, coord, rotb, N);
  wmma_gemm_bf16<128, 1><<<N / 16, blk, 0, stream>>>(rotb, PTbf, misc + ZBIAS, qrope);
  // k = rot(k_raw @ P^T) @ P
  wmma_gemm_bf16<128, 1><<<N / 16, blk, 0, stream>>>(kraw, Pbf, misc + ZBIAS, xr);
  rope_rotate<<<(N * 64) / 256, blk, 0, stream>>>(xr, coord, rotb, N);
  wmma_gemm_bf16<128, 1><<<N / 16, blk, 0, stream>>>(rotb, PTbf, misc + ZBIAS, krope);
  // gate
  gate_in_build<<<(N * 2 * NC) / 256, blk, 0, stream>>>(qraw, misc, offset, B, ginb, N);
  wmma_gemm_bf16<256, 1><<<N / 16, blk, 0, stream>>>(ginb, Wg1T, bg1, hbuf);
  bn_stats<<<256, blk, 0, stream>>>(hbuf, N, NC, NC, misc + H_SUM, misc + H_SQ);
  gate_final<<<(N * NG + 255) / 256, blk, 0, stream>>>(hbuf, misc, gg, bng, Wg2, bg2, gatev, N);
  // geom + attention + output
  geom_pass1<<<(N * NK * NG) / 256, blk, 0, stream>>>(qrope, krope, refidx, Ww1, bw1,
                                                      hbuf, misc, N);
  geom_final<<<N, dim3(128), 0, stream>>>(hbuf, misc, gw, bnw, Ww2, bw2, gatev, refidx,
                                          qkv, (float*)d_out, N);
}
